// S2sTarget_42417097016350
// MI455X (gfx1250) — matrix operand order repaired and checked by reading, not compiled, not run on hardware
//
#include <hip/hip_runtime.h>
#include <hip/hip_bf16.h>

// ---------------- problem constants ----------------
constexpr int kVocab = 32000;
constexpr int kEmb   = 512;
constexpr int kHid   = 512;
constexpr int kG     = 4 * kHid;   // 2048 gate columns
constexpr int kBB    = 16;         // batch
constexpr int kT     = 128;        // target length
constexpr int kSenc  = 64;
constexpr int kRows  = kBB * kT;   // 2048 (b*T+t) rows
constexpr int kNvt   = 256;        // vocab tile width
constexpr int kNtv   = kVocab / kNvt; // 125 vocab tiles
constexpr int kNblk  = 16;         // persistent LSTM workgroups

// ---------------- WMMA types ----------------
typedef __attribute__((ext_vector_type(16))) _Float16 v16h;
typedef __attribute__((ext_vector_type(8)))  _Float16 v8h;
typedef __attribute__((ext_vector_type(8)))  float    v8f;

union V16H { v16h v; v8h h[2]; };

// A-matrix fragment (16xK tile, 16-bit): lane l -> row m0+(l&15),
// halves [kb..kb+7] and [kb+16..kb+23], kb = (l<16)?0:8   (ISA 7.12.2)
__device__ inline v16h load_a_frag(const _Float16* __restrict__ A, int lda,
                                   int m0, int k0, int lane) {
  int m  = m0 + (lane & 15);
  int kb = (lane >> 4) ? 8 : 0;
  const _Float16* p = A + (long)m * lda + k0 + kb;
  V16H u;
  u.h[0] = *(const v8h*)(p);        // K = kb .. kb+7
  u.h[1] = *(const v8h*)(p + 16);   // K = kb+16 .. kb+23
  return u.v;
}

// B-matrix fragment (Kx16 tile, 16-bit), B column n == row n of W[N][K]:
// lane l -> col n0+(l&15), K [kb2..kb2+7] and [kb2+16..kb2+23], kb2 = (l<16)?0:8
__device__ inline v16h load_b_frag(const _Float16* __restrict__ W, int ldw,
                                   int n0, int k0, int lane) {
  int n   = n0 + (lane & 15);
  int kb2 = (lane >> 4) ? 8 : 0;
  const _Float16* p = W + (long)n * ldw + k0 + kb2;
  V16H u;
  u.h[0] = *(const v8h*)(p);
  u.h[1] = *(const v8h*)(p + 16);
  return u.v;
}

__device__ inline v8f wmma_f16(v16h a, v16h b, v8f c) {
  return __builtin_amdgcn_wmma_f32_16x16x32_f16(false, a, false, b,
                                                (short)0, c, false, false);
}

__device__ inline float sigmoidf_(float x) { return 1.0f / (1.0f + __expf(-x)); }

// ---------------- grid barrier (16 co-resident WGs) ----------------
__device__ inline void grid_barrier(int* cnt, volatile int* gen, int tid, int nblk) {
  __syncthreads();
  if (tid == 0) {
    __threadfence();
    int g0 = *gen;
    if (atomicAdd(cnt, 1) == nblk - 1) {
      *cnt = 0;
      __threadfence();
      atomicAdd((int*)gen, 1);
    } else {
      while (*gen == g0) { __builtin_amdgcn_s_sleep(2); }
      __threadfence();
    }
  }
  __syncthreads();
}

// ---------------- kernels ----------------
__global__ void k_init(int* bar) {
  if (threadIdx.x < 2 && blockIdx.x == 0) bar[threadIdx.x] = 0;
}

__global__ void k_f32_to_f16(const float* __restrict__ src,
                             _Float16* __restrict__ dst, int n) {
  int i = blockIdx.x * blockDim.x + threadIdx.x;
  int stride = gridDim.x * blockDim.x;
  for (; i < n; i += stride) dst[i] = (_Float16)src[i];
}

// emb_h row order = t*B + b (matches x_proj / lstm consumption)
__global__ void k_gather_emb(const float* __restrict__ embedding,
                             const int* __restrict__ tgt,
                             _Float16* __restrict__ emb_h) {
  int row = blockIdx.x;            // row = t*kBB + b
  int t = row / kBB, b = row % kBB;
  int tok = tgt[b * kT + t];
  const float* src = embedding + (long)tok * kEmb;
  _Float16* dst = emb_h + (long)row * kEmb;
  for (int e = threadIdx.x; e < kEmb; e += blockDim.x)
    dst[e] = (_Float16)src[e];
}

// x_proj[t*B+b][g] = emb @ W_ih^T + b_ih + b_hh    (M=2048,N=2048,K=512)
__global__ __launch_bounds__(256)
void k_xproj(const _Float16* __restrict__ emb_h,
             const _Float16* __restrict__ Wih_h,
             const float* __restrict__ b_ih, const float* __restrict__ b_hh,
             float* __restrict__ xp) {
  int lane = threadIdx.x & 31;
  int wave = threadIdx.x >> 5;
  int m0 = blockIdx.y * 16;
  int n0 = (blockIdx.x * 8 + wave) * 16;
  v8f acc = {};
  for (int k0 = 0; k0 < kEmb; k0 += 32) {
    v16h a = load_a_frag(emb_h, kEmb, m0, k0, lane);
    v16h b = load_b_frag(Wih_h, kEmb, n0, k0, lane);
    acc = wmma_f16(a, b, acc);
  }
  int n = n0 + (lane & 15);
  float bias = b_ih[n] + b_hh[n];
  int mbase = m0 + ((lane >> 4) ? 8 : 0);
#pragma unroll
  for (int r = 0; r < 8; ++r)
    xp[(long)(mbase + r) * kG + n] = acc[r] + bias;
}

// Persistent LSTM: 16 WGs, WG w owns HID cols [32w,32w+32).
// h double-buffered in global (f16), staged through LDS each step; c in LDS.
__global__ __launch_bounds__(256)
void k_lstm(const float* __restrict__ memory_bank,
            const _Float16* __restrict__ Whh_h,
            const float* __restrict__ xp,
            _Float16* __restrict__ h_buf,      // 2 * 16 * 512 f16
            _Float16* __restrict__ hs_h,       // [b*T+t][512] f16
            int* bar_cnt, int* bar_gen) {
  __shared__ _Float16 h_s[kBB * kHid];   // 16 KB: full h, shared by all 8 waves
  __shared__ float c_s[16][32];
  __shared__ float g_s[16][128];   // [b][ i(0-31) f(32-63) g(64-95) o(96-127) ]
  const int w = blockIdx.x;
  const int tid = threadIdx.x;
  const int lane = tid & 31, wave = tid >> 5;

  // init c and h0 from memory_bank[:, S_ENC-1, :]
  for (int i = tid; i < 16 * 32; i += 256) {
    int b = i >> 5, col = i & 31;
    float v = memory_bank[((long)b * kSenc + (kSenc - 1)) * kHid + 32 * w + col];
    c_s[b][col] = v;
    h_buf[(long)b * kHid + 32 * w + col] = (_Float16)v;
  }
  grid_barrier(bar_cnt, bar_gen, tid, kNblk);

  const int gate = wave >> 1;              // 0=i 1=f 2=g 3=o
  const int sub  = (wave & 1) * 16;
  const int gcol = gate * kHid + 32 * w + sub;   // 16 gate columns per wave

  for (int t = 0; t < kT; ++t) {
    const _Float16* h_in  = h_buf + (long)(t & 1) * (kBB * kHid);
    _Float16*       h_out = h_buf + (long)((t + 1) & 1) * (kBB * kHid);

    // stage full h into LDS once (instead of 8 redundant global streams)
    {
      const v8h* src = (const v8h*)h_in;
      v8h* dst = (v8h*)h_s;
      for (int i = tid; i < (kBB * kHid) / 8; i += 256) dst[i] = src[i];
    }
    // prefetch next step's xp addend while we compute (global_prefetch path)
    if (t + 1 < kT)
      __builtin_prefetch(&xp[((long)(t + 1) * kBB) * kG + gcol], 0, 1);
    __syncthreads();

    v8f acc = {};
    for (int k0 = 0; k0 < kHid; k0 += 32) {
      v16h a = load_a_frag(h_s, kHid, 0, k0, lane);           // M = batch(16)
      v16h b = load_b_frag(Whh_h, kHid, gcol, k0, lane);
      acc = wmma_f16(a, b, acc);
    }
    int n = lane & 15;
    int bbase = (lane >> 4) ? 8 : 0;
#pragma unroll
    for (int r = 0; r < 8; ++r) {
      int b = bbase + r;
      g_s[b][gate * 32 + sub + n] =
          acc[r] + xp[((long)t * kBB + b) * kG + gcol + n];
    }
    __syncthreads();

    for (int i = tid; i < 16 * 32; i += 256) {
      int b = i >> 5, col = i & 31;
      float ig = sigmoidf_(g_s[b][col]);
      float fg = sigmoidf_(g_s[b][32 + col]);
      float gg = tanhf(g_s[b][64 + col]);
      float og = sigmoidf_(g_s[b][96 + col]);
      float c = fg * c_s[b][col] + ig * gg;
      c_s[b][col] = c;
      float h = og * tanhf(c);
      int hc = 32 * w + col;
      h_out[(long)b * kHid + hc] = (_Float16)h;
      hs_h[((long)b * kT + t) * kHid + hc] = (_Float16)h;
    }
    grid_barrier(bar_cnt, bar_gen, tid, kNblk);   // also fences h_s reuse
  }
}

// logits tile (16 rows x 256 vocab) + fused online-softmax partials.
// A-tile (hs rows) staged once in LDS and shared by all 8 waves.
// partials[row][tile] = { max, sumexp(rel max), argmax_col, target_logit or -1e30 }
__global__ __launch_bounds__(256)
void k_logits(const _Float16* __restrict__ hs_h,
              const _Float16* __restrict__ Wout_h,
              const float* __restrict__ b_out,
              const int* __restrict__ tgt,
              float* __restrict__ partials) {
  __shared__ _Float16 a_s[16 * kHid];      // 16 KB A tile
  __shared__ float tile[16][kNvt + 1];
  __shared__ float rmax[16][16];
  __shared__ int   ridx[16][16];
  __shared__ float rsum[16][16];
  const int tid = threadIdx.x, lane = tid & 31, wave = tid >> 5;
  const int m0 = blockIdx.y * 16;
  const int v0 = blockIdx.x * kNvt;

  // cooperative stage of hs rows m0..m0+15 into LDS
  {
    const v8h* src = (const v8h*)(hs_h + (long)m0 * kHid);
    v8h* dst = (v8h*)a_s;
    for (int i = tid; i < (16 * kHid) / 8; i += 256) dst[i] = src[i];
  }
  __syncthreads();

#pragma unroll
  for (int s = 0; s < 2; ++s) {
    int nt = wave * 2 + s;
    int n0 = v0 + nt * 16;
    v8f acc = {};
    for (int k0 = 0; k0 < kHid; k0 += 32) {
      v16h a = load_a_frag(a_s, kHid, 0, k0, lane);
      v16h b = load_b_frag(Wout_h, kHid, n0, k0, lane);
      acc = wmma_f16(a, b, acc);
    }
    int n = lane & 15;
    float bias = b_out[n0 + n];
    int rbase = (lane >> 4) ? 8 : 0;
#pragma unroll
    for (int r = 0; r < 8; ++r)
      tile[rbase + r][nt * 16 + n] = acc[r] + bias;
  }
  __syncthreads();

  const int row = tid >> 4;   // 0..15
  const int q   = tid & 15;
  float mx = -1e30f; int ai = 0;
  for (int c = q; c < kNvt; c += 16) {
    float v = tile[row][c];
    if (v > mx) { mx = v; ai = c; }
  }
  rmax[row][q] = mx; ridx[row][q] = ai;
  __syncthreads();
  if (q == 0) {
    float M = rmax[row][0]; int I = ridx[row][0];
    for (int j = 1; j < 16; ++j)
      if (rmax[row][j] > M) { M = rmax[row][j]; I = ridx[row][j]; }
    rmax[row][0] = M; ridx[row][0] = I;
  }
  __syncthreads();
  float M = rmax[row][0];
  float se = 0.0f;
  for (int c = q; c < kNvt; c += 16) se += __expf(tile[row][c] - M);
  rsum[row][q] = se;
  __syncthreads();
  if (q == 0) {
    float S = 0.0f;
    for (int j = 0; j < 16; ++j) S += rsum[row][j];
    int m = m0 + row;           // global row = b*T+t  (== tgt flat index)
    int tok = tgt[m];
    float tl = -1e30f;
    if (tok >= v0 && tok < v0 + kNvt) tl = tile[row][tok - v0];
    float* p = partials + ((long)m * kNtv + blockIdx.x) * 4;
    p[0] = M;
    p[1] = S;
    p[2] = (float)(v0 + ridx[row][0]);
    p[3] = tl;
  }
}

// per-row combine of 125 partial tiles -> {mask*nll, mask*correct, mask}
__global__ void k_reduce(const float* __restrict__ partials,
                         const int* __restrict__ tgt,
                         float* __restrict__ rowres) {
  int m = blockIdx.x * blockDim.x + threadIdx.x;
  if (m >= kRows) return;
  const float* p = partials + (long)m * kNtv * 4;
  float M = -1e30f;
  for (int i = 0; i < kNtv; ++i) M = fmaxf(M, p[i * 4 + 0]);
  float S = 0.0f, amv = -1e30f, tl = -1e30f; int amx = 0;
  for (int i = 0; i < kNtv; ++i) {
    S += p[i * 4 + 1] * __expf(p[i * 4 + 0] - M);
    if (p[i * 4 + 0] > amv) { amv = p[i * 4 + 0]; amx = (int)p[i * 4 + 2]; }
    tl = fmaxf(tl, p[i * 4 + 3]);
  }
  float logp_t = tl - M - __logf(S);
  int tok = tgt[m];
  float mask = (tok > 0) ? 1.0f : 0.0f;
  rowres[m * 3 + 0] = mask * (-logp_t);
  rowres[m * 3 + 1] = mask * ((amx == tok) ? 1.0f : 0.0f);
  rowres[m * 3 + 2] = mask;
}

// deterministic single-thread final sum -> (loss, correct, denominator)
__global__ void k_finalize(const float* __restrict__ rowres, float* __restrict__ out) {
  if (threadIdx.x != 0 || blockIdx.x != 0) return;
  float l = 0.0f, c = 0.0f, d = 0.0f;
  for (int m = 0; m < kRows; ++m) {
    l += rowres[m * 3 + 0];
    c += rowres[m * 3 + 1];
    d += rowres[m * 3 + 2];
  }
  out[0] = l / (d + 1e-6f);
  out[1] = c;
  out[2] = d;
}

// ---------------- host launch ----------------
extern "C" void kernel_launch(void* const* d_in, const int* in_sizes, int n_in,
                              void* d_out, int out_size, void* d_ws, size_t ws_size,
                              hipStream_t stream) {
  const float* memory_bank = (const float*)d_in[0];
  const int*   tgt         = (const int*)d_in[1];
  const float* embedding   = (const float*)d_in[2];
  const float* W_ih        = (const float*)d_in[3];
  const float* W_hh        = (const float*)d_in[4];
  const float* b_ih        = (const float*)d_in[5];
  const float* b_hh        = (const float*)d_in[6];
  const float* W_out       = (const float*)d_in[7];
  const float* b_out       = (const float*)d_in[8];
  float* out = (float*)d_out;
  char*  ws  = (char*)d_ws;

  // workspace layout (bytes)
  constexpr size_t o_emb    = 0;                                    // 2 MB f16
  constexpr size_t o_wih    = o_emb    + (size_t)kRows * kEmb * 2;  // 2 MB f16
  constexpr size_t o_whh    = o_wih    + (size_t)kG * kEmb * 2;     // 2 MB f16
  constexpr size_t o_wout   = o_whh    + (size_t)kG * kHid * 2;     // 32.8 MB f16
  constexpr size_t o_xp     = o_wout   + (size_t)kVocab * kHid * 2; // 16.8 MB f32
  constexpr size_t o_hs     = o_xp     + (size_t)kT * kBB * kG * 4; // 2 MB f16
  constexpr size_t o_hbuf   = o_hs     + (size_t)kRows * kHid * 2;  // 32 KB f16
  constexpr size_t o_part   = o_hbuf   + (size_t)2 * kBB * kHid * 2;// 4.1 MB f32
  constexpr size_t o_rowres = o_part   + (size_t)kRows * kNtv * 4 * 4;
  constexpr size_t o_bar    = o_rowres + (size_t)kRows * 3 * 4;

  _Float16* emb_h   = (_Float16*)(ws + o_emb);
  _Float16* Wih_h   = (_Float16*)(ws + o_wih);
  _Float16* Whh_h   = (_Float16*)(ws + o_whh);
  _Float16* Wout_h  = (_Float16*)(ws + o_wout);
  float*    xp      = (float*)(ws + o_xp);
  _Float16* hs_h    = (_Float16*)(ws + o_hs);
  _Float16* h_buf   = (_Float16*)(ws + o_hbuf);
  float*    part    = (float*)(ws + o_part);
  float*    rowres  = (float*)(ws + o_rowres);
  int*      bar     = (int*)(ws + o_bar);

  k_init<<<1, 64, 0, stream>>>(bar);
  k_f32_to_f16<<<1024, 256, 0, stream>>>(W_ih,  Wih_h,  kG * kEmb);
  k_f32_to_f16<<<1024, 256, 0, stream>>>(W_hh,  Whh_h,  kG * kHid);
  k_f32_to_f16<<<4096, 256, 0, stream>>>(W_out, Wout_h, kVocab * kHid);
  k_gather_emb<<<kRows, 256, 0, stream>>>(embedding, tgt, emb_h);

  // x_proj GEMM: grid.x covers N (2048 / (8 waves * 16)), grid.y covers M
  k_xproj<<<dim3(kG / 128, kRows / 16), 256, 0, stream>>>(emb_h, Wih_h, b_ih, b_hh, xp);

  // persistent LSTM scan
  k_lstm<<<kNblk, 256, 0, stream>>>(memory_bank, Whh_h, xp, h_buf, hs_h,
                                    bar, bar + 1);

  // fused logits + online softmax partials
  k_logits<<<dim3(kNtv, kRows / 16), 256, 0, stream>>>(hs_h, Wout_h, b_out, tgt, part);

  k_reduce<<<kRows / 256, 256, 0, stream>>>(part, tgt, rowres);
  k_finalize<<<1, 1, 0, stream>>>(rowres, out);
}
